// Mix_82360292868539
// MI455X (gfx1250) — matrix operand order repaired and checked by reading, not compiled, not run on hardware
//
#include <hip/hip_runtime.h>
#include <hip/hip_bf16.h>

typedef __attribute__((ext_vector_type(8)))  float  v8f;
typedef __attribute__((ext_vector_type(16))) __bf16 v16bf;

#define N_DIM 8192
#define D_DIM 64
#define BM 128
#define BN 64
#define LDT 72              // LDS row stride in bf16 elems (64 + 8 pad, 144B = 9*16B)
#define NBINS 2048
#define HIST_SCALE 4.0f     // bins per unit pdist; covers [0, 512)
#define R_COEF 0.1f

// ---------- bf16 helpers (bit-exact RNE, no reliance on native __bf16 casts) ----------
__device__ __forceinline__ unsigned short f2bf(float f) {
    unsigned int u = __float_as_uint(f);
    u += 0x7fffu + ((u >> 16) & 1u);
    return (unsigned short)(u >> 16);
}
__device__ __forceinline__ float bf2f(unsigned short h) {
    return __uint_as_float(((unsigned int)h) << 16);
}

// ---------- WMMA fragment loads from LDS (layouts per ISA 7.12.2, wave32) ----------
// A (16x32 bf16): lane l(0-15)=row M=l holds K {0..7,16..23}; lane l+16 holds K {8..15,24..31}
__device__ __forceinline__ v16bf load_a_frag(const unsigned short* s, int row_base, int kk, int lane) {
    const int l  = lane & 15;
    const int hi = lane >> 4;                       // 0 or 1
    const unsigned short* p = s + (row_base + l) * LDT + kk + hi * 8;
    union { v16bf v; uint4 u[2]; } r;
    r.u[0] = *(const uint4*)(p);                    // K = kk+hi*8 .. +7      -> VGPR0..3
    r.u[1] = *(const uint4*)(p + 16);               // K = kk+16+hi*8 .. +7   -> VGPR4..7
    return r.v;
}
// B (32x16 bf16): lane n(0-15)=col N=n holds K {0..7,16..23}; lane n+16 holds K {8..15,24..31}
__device__ __forceinline__ v16bf load_b_frag(const unsigned short* s, int col_base, int kk, int lane) {
    const int n  = lane & 15;
    const int kh = lane >> 4;                       // 0 or 1
    const unsigned short* p = s + (col_base + n) * LDT + kk + kh * 8;
    union { v16bf v; uint4 u[2]; } r;
    r.u[0] = *(const uint4*)(p);
    r.u[1] = *(const uint4*)(p + 16);
    return r.v;
}

// ---------- global f32 -> LDS split-bf16 staging ----------
__device__ __forceinline__ void stage_tile(const float* __restrict__ src, int row0, int rows,
                                           unsigned short* dh, unsigned short* dl, int tid) {
    const float4* sv = (const float4*)src + (size_t)row0 * (D_DIM / 4);
    for (int i = tid; i < rows * (D_DIM / 4); i += 256) {
        const int r = i >> 4;
        const int c = (i & 15) << 2;
        float4 v = sv[i];
        const int o = r * LDT + c;
        float f; unsigned short h;
        f = v.x; h = f2bf(f); dh[o + 0] = h; dl[o + 0] = f2bf(f - bf2f(h));
        f = v.y; h = f2bf(f); dh[o + 1] = h; dl[o + 1] = f2bf(f - bf2f(h));
        f = v.z; h = f2bf(f); dh[o + 2] = h; dl[o + 2] = f2bf(f - bf2f(h));
        f = v.w; h = f2bf(f); dh[o + 3] = h; dl[o + 3] = f2bf(f - bf2f(h));
    }
}

// ---------- split-bf16 GEMM core: inner = hi*hi + hi*lo + lo*hi (~f32 accuracy) ----------
__device__ __forceinline__ void mma_tile(const unsigned short* sxh, const unsigned short* sxl,
                                         const unsigned short* syh, const unsigned short* syl,
                                         int wave_m, int wave_n, int lane, v8f acc[2][2]) {
#pragma unroll
    for (int kk = 0; kk < D_DIM; kk += 32) {
        v16bf ah[2], al[2], bh[2], bl[2];
#pragma unroll
        for (int mt = 0; mt < 2; ++mt) {
            ah[mt] = load_a_frag(sxh, wave_m * 32 + mt * 16, kk, lane);
            al[mt] = load_a_frag(sxl, wave_m * 32 + mt * 16, kk, lane);
        }
#pragma unroll
        for (int nt = 0; nt < 2; ++nt) {
            bh[nt] = load_b_frag(syh, wave_n * 32 + nt * 16, kk, lane);
            bl[nt] = load_b_frag(syl, wave_n * 32 + nt * 16, kk, lane);
        }
#pragma unroll
        for (int mt = 0; mt < 2; ++mt)
#pragma unroll
            for (int nt = 0; nt < 2; ++nt) {
                acc[mt][nt] = __builtin_amdgcn_wmma_f32_16x16x32_bf16(
                    false, ah[mt], false, bh[nt], (short)0, acc[mt][nt], false, false);
                acc[mt][nt] = __builtin_amdgcn_wmma_f32_16x16x32_bf16(
                    false, ah[mt], false, bl[nt], (short)0, acc[mt][nt], false, false);
                acc[mt][nt] = __builtin_amdgcn_wmma_f32_16x16x32_bf16(
                    false, al[mt], false, bh[nt], (short)0, acc[mt][nt], false, false);
            }
    }
}

// ---------- kernel 1: per-row squared norms ----------
__global__ void __launch_bounds__(256) row_norms_kernel(const float* __restrict__ x,
                                                        const float* __restrict__ y,
                                                        float* __restrict__ xs,
                                                        float* __restrict__ ys) {
    const int i = blockIdx.x * 256 + threadIdx.x;   // 0..8191
    const float4* xv = (const float4*)x + (size_t)i * 16;
    const float4* yv = (const float4*)y + (size_t)i * 16;
    float sx = 0.f, sy = 0.f;
#pragma unroll
    for (int c = 0; c < 16; ++c) {
        float4 a = xv[c]; sx += a.x * a.x + a.y * a.y + a.z * a.z + a.w * a.w;
        float4 b = yv[c]; sy += b.x * b.x + b.y * b.y + b.z * b.z + b.w * b.w;
    }
    xs[i] = sx;
    ys[i] = sy;
}

// ---------- kernel 2: zero histogram ----------
__global__ void __launch_bounds__(256) zero_hist_kernel(unsigned int* __restrict__ hist) {
    const int i = blockIdx.x * 256 + threadIdx.x;
    if (i < NBINS) hist[i] = 0u;
}

// ---------- kernel 3: pdist histogram pass (recomputes GEMM via WMMA) ----------
__global__ void __launch_bounds__(256) hist_kernel(const float* __restrict__ x,
                                                   const float* __restrict__ y,
                                                   const float* __restrict__ xs,
                                                   const float* __restrict__ ys,
                                                   unsigned int* __restrict__ hist) {
    __shared__ __align__(16) unsigned short sxh[BM * LDT], sxl[BM * LDT];
    __shared__ __align__(16) unsigned short syh[BN * LDT], syl[BN * LDT];
    __shared__ float s_xs[BM], s_ys[BN];
    __shared__ unsigned int lhist[NBINS];

    const int tid = threadIdx.x, lane = tid & 31, w = tid >> 5;
    const int wave_m = w >> 1, wave_n = w & 1;

    for (int i = tid; i < NBINS; i += 256) lhist[i] = 0u;

    const int ntiles_n = N_DIM / BN;                 // 128
    const int ntiles   = (N_DIM / BM) * ntiles_n;    // 8192
    for (int t = blockIdx.x; t < ntiles; t += gridDim.x) {
        const int row0 = (t / ntiles_n) * BM;
        const int col0 = (t % ntiles_n) * BN;
        __syncthreads();
        stage_tile(x, row0, BM, sxh, sxl, tid);
        stage_tile(y, col0, BN, syh, syl, tid);
        if (tid < BM) s_xs[tid] = xs[row0 + tid];
        if (tid < BN) s_ys[tid] = ys[col0 + tid];
        __syncthreads();

        v8f acc[2][2] = {};
        mma_tile(sxh, sxl, syh, syl, wave_m, wave_n, lane, acc);

#pragma unroll
        for (int mt = 0; mt < 2; ++mt)
#pragma unroll
            for (int nt = 0; nt < 2; ++nt)
#pragma unroll
                for (int r = 0; r < 8; ++r) {
                    const int rl = wave_m * 32 + mt * 16 + r + ((lane >> 4) << 3);
                    const int cl = wave_n * 32 + nt * 16 + (lane & 15);
                    const float a = acc[mt][nt][r];
                    const float pd = s_xs[rl] + s_ys[cl] - 2.0f * a;
                    int b = (int)(pd * HIST_SCALE);
                    b = b < 0 ? 0 : (b > NBINS - 1 ? NBINS - 1 : b);
                    atomicAdd(&lhist[b], 1u);
                }
    }
    __syncthreads();
    for (int i = tid; i < NBINS; i += 256)
        if (lhist[i]) atomicAdd(&hist[i], lhist[i]);
}

// ---------- kernel 4: median -> scale = ln(N+1)/median = 1/(2*sigma) ----------
__global__ void median_kernel(const unsigned int* __restrict__ hist,
                              float* __restrict__ scale_out) {
    if (threadIdx.x == 0) {
        const unsigned long long k1 = 33554432ull;   // rank N*M/2   (1-based)
        const unsigned long long k2 = 33554433ull;   // rank N*M/2+1
        unsigned long long cum = 0;
        float m1 = -1.0f, m2 = -1.0f;
        for (int b = 0; b < NBINS; ++b) {
            cum += hist[b];
            const float ctr = (b + 0.5f) * (1.0f / HIST_SCALE);
            if (m1 < 0.0f && cum >= k1) m1 = ctr;
            if (m2 < 0.0f && cum >= k2) { m2 = ctr; break; }
        }
        float med = 0.5f * (m1 + m2);
        if (!(med > 1e-6f)) med = 1e-6f;
        scale_out[0] = logf((float)(N_DIM + 1)) / med;   // 1/(2*sigma)
    }
}

// ---------- kernel 5: fused GEMM + kernel-matrix epilogue ----------
__global__ void __launch_bounds__(256) kxy_kernel(const float* __restrict__ x,
                                                  const float* __restrict__ y,
                                                  const float* __restrict__ xs,
                                                  const float* __restrict__ ys,
                                                  const float* __restrict__ scale_p,
                                                  float* __restrict__ out) {
    __shared__ __align__(16) unsigned short sxh[BM * LDT], sxl[BM * LDT];
    __shared__ __align__(16) unsigned short syh[BN * LDT], syl[BN * LDT];
    __shared__ float s_xs[BM], s_ys[BN];

    const int tid = threadIdx.x, lane = tid & 31, w = tid >> 5;
    const int wave_m = w >> 1, wave_n = w & 1;
    const int row0 = blockIdx.y * BM;
    const int col0 = blockIdx.x * BN;

    stage_tile(x, row0, BM, sxh, sxl, tid);
    stage_tile(y, col0, BN, syh, syl, tid);
    if (tid < BM) s_xs[tid] = xs[row0 + tid];
    if (tid < BN) s_ys[tid] = ys[col0 + tid];
    __syncthreads();

    v8f acc[2][2] = {};
    mma_tile(sxh, sxl, syh, syl, wave_m, wave_n, lane, acc);

    const float sc = scale_p[0];
#pragma unroll
    for (int mt = 0; mt < 2; ++mt)
#pragma unroll
        for (int nt = 0; nt < 2; ++nt)
#pragma unroll
            for (int r = 0; r < 8; ++r) {
                const int rl = wave_m * 32 + mt * 16 + r + ((lane >> 4) << 3);
                const int cl = wave_n * 32 + nt * 16 + (lane & 15);
                const float a = acc[mt][nt][r];
                const float pd = s_xs[rl] + s_ys[cl] - 2.0f * a;
                const float v = __expf(-pd * sc) + R_COEF * a * a;  // C == 0
                // 256MB streaming output: non-temporal store, don't pollute L2
                __builtin_nontemporal_store(v, &out[(size_t)(row0 + rl) * N_DIM + (col0 + cl)]);
            }
}

extern "C" void kernel_launch(void* const* d_in, const int* in_sizes, int n_in,
                              void* d_out, int out_size, void* d_ws, size_t ws_size,
                              hipStream_t stream) {
    const float* x = (const float*)d_in[0];
    const float* y = (const float*)d_in[1];
    float* out = (float*)d_out;

    // workspace layout
    unsigned int* hist = (unsigned int*)d_ws;                      // 2048 * 4  =  8KB
    float* xs    = (float*)((char*)d_ws + 8192);                   // 8192 * 4  = 32KB
    float* ys    = (float*)((char*)d_ws + 8192 + 32768);           // 8192 * 4  = 32KB
    float* scale = (float*)((char*)d_ws + 8192 + 65536);           // 4B

    zero_hist_kernel<<<(NBINS + 255) / 256, 256, 0, stream>>>(hist);
    row_norms_kernel<<<N_DIM / 256, 256, 0, stream>>>(x, y, xs, ys);
    hist_kernel<<<512, 256, 0, stream>>>(x, y, xs, ys, hist);
    median_kernel<<<1, 32, 0, stream>>>(hist, scale);
    kxy_kernel<<<dim3(N_DIM / BN, N_DIM / BM), 256, 0, stream>>>(x, y, xs, ys, scale, out);
}
